// causal_multihead_self_attention_82291573392177
// MI455X (gfx1250) — compile-verified, matrix operand order checked by reading, not run on hardware
//
#include <hip/hip_runtime.h>
#include <hip/hip_bf16.h>

typedef __bf16 v16bf __attribute__((ext_vector_type(16)));
typedef __bf16 v8bf  __attribute__((ext_vector_type(8)));
typedef float  v8f   __attribute__((ext_vector_type(8)));

#define WMMA_BF16(A, B, C) \
    __builtin_amdgcn_wmma_f32_16x16x32_bf16(false, (A), false, (B), (short)0, (C), false, false)

// f32 -> bf16, round to nearest even (used only in elementwise converts/epilogues)
__device__ __forceinline__ __bf16 f2bf(float f) {
    unsigned u = __builtin_bit_cast(unsigned, f);
    u += 0x7fffu + ((u >> 16) & 1u);
    unsigned short h = (unsigned short)(u >> 16);
    return __builtin_bit_cast(__bf16, h);
}

// 16-bit A/B fragment for wmma 16x16x32, built from a contiguous 32-element
// bf16 row. Mapping k = (v&3)*2 + (v>>2)*16 + half*8 means the fragment is
// exactly two contiguous 16-byte chunks: [half*8 .. +8) and [16+half*8 .. +8).
// => two b128 loads, zero data shuffling.
__device__ __forceinline__ v16bf load_frag(const __bf16* row, int half) {
    const v8bf lo = *(const v8bf*)(row + half * 8);
    const v8bf hi = *(const v8bf*)(row + 16 + half * 8);
    return __builtin_shufflevector(lo, hi, 0, 1, 2, 3, 4, 5, 6, 7,
                                           8, 9, 10, 11, 12, 13, 14, 15);
}

__device__ __forceinline__ v8f zero_v8f() {
    v8f c;
#pragma unroll
    for (int i = 0; i < 8; ++i) c[i] = 0.0f;
    return c;
}

// -------------------------------------------------------------------------
// Kernel 0: elementwise f32 -> bf16 convert (memory bound, done once).
// Each thread converts 4 elements.
// -------------------------------------------------------------------------
__global__ __launch_bounds__(256) void cvt_bf16_kernel(
    const float* __restrict__ in, __bf16* __restrict__ out) {
    const size_t i = ((size_t)blockIdx.x * 256 + threadIdx.x) * 4;
    const float4 f = *(const float4*)(in + i);
    v8bf* o = (v8bf*)0;  // silence unused-type warnings pattern
    (void)o;
    __bf16 b0 = f2bf(f.x), b1 = f2bf(f.y), b2 = f2bf(f.z), b3 = f2bf(f.w);
    __bf16 tmp[4] = {b0, b1, b2, b3};
    *(uint2*)(out + i) = *(const uint2*)tmp;
}

// -------------------------------------------------------------------------
// Kernel 1: qkv = x @ Wqkv^T, fused RoPE on q/k, bf16 pack:
//   Q,K -> [B,H,S,hd]   V -> [B,H,hd,S] (transposed)
// One wave computes a 16x64 tile of the [4096 x 3072] output
// (1 A-frag load feeds 4 WMMAs per K-step).
// -------------------------------------------------------------------------
__global__ __launch_bounds__(128) void qkv_rope_kernel(
    const __bf16* __restrict__ xb, const int* __restrict__ pos,
    const __bf16* __restrict__ wb,
    __bf16* __restrict__ Qb, __bf16* __restrict__ Kb, __bf16* __restrict__ Vt) {
    const int D = 1024, S = 2048;
    const int lane = threadIdx.x & 31, wid = threadIdx.x >> 5;
    const int half = lane >> 4, rown = lane & 15;
    const int w  = blockIdx.x * 4 + wid;
    const int nt = w % 48;    // 3072/64 column tiles
    const int mt = w / 48;    // 4096/16 row tiles (flattened B*S)

    const __bf16* xrow = xb + (size_t)(mt * 16 + rown) * D;           // A row
    const __bf16* wrow = wb + (size_t)(nt * 64 + rown) * D;           // B col base

    v8f acc[4];
#pragma unroll
    for (int t = 0; t < 4; ++t) acc[t] = zero_v8f();

#pragma unroll 4
    for (int kc = 0; kc < 32; ++kc) {
        const v16bf a = load_frag(xrow + kc * 32, half);
#pragma unroll
        for (int sub = 0; sub < 4; ++sub) {
            const v16bf b = load_frag(wrow + (size_t)sub * 16 * D + kc * 32, half);
            acc[sub] = WMMA_BF16(a, b, acc[sub]);
        }
    }

    // Tile covers e in [nt*64, nt*64+64): exactly one head, one section.
    const int e0  = nt * 64;
    const int sec = e0 >> 10;            // 0=q, 1=k, 2=v
    const int hh  = (e0 & 1023) >> 6;    // head

#pragma unroll
    for (int sub = 0; sub < 4; ++sub) {
        const int d = sub * 16 + rown;   // dim within head (0..63)
        if (sec < 2) {
            const int i = d >> 1;
            // inv_freq = 10000^(-2i/64) = exp(-i * ln(1e4)/32)
            const float invf = __expf(-0.28782313662425572f * (float)i);
            __bf16* dst = (sec == 0) ? Qb : Kb;
#pragma unroll
            for (int r = 0; r < 8; ++r) {
                const int g  = mt * 16 + r + 8 * half;
                const int b_ = g >> 11, s = g & 2047;
                const float p   = (float)pos[s];
                const float ang = p * invf;
                const float cv = __cosf(ang), sv = __sinf(ang);
                const float v    = acc[sub][r];
                const float part = __shfl_xor(v, 1, 32);
                const float nv = (d & 1) ? (part * sv + v * cv)
                                         : (v * cv - part * sv);
                dst[((size_t)(b_ * 16 + hh) * S + s) * 64 + d] = f2bf(nv);
            }
        } else {
#pragma unroll
            for (int r = 0; r < 8; ++r) {
                const int g  = mt * 16 + r + 8 * half;
                const int b_ = g >> 11, s = g & 2047;
                Vt[((size_t)(b_ * 16 + hh) * 64 + d) * S + s] = f2bf(acc[sub][r]);
            }
        }
    }
}

// -------------------------------------------------------------------------
// Kernel 2: flash-style causal attention. One wave per (b,h,16-row q tile).
// Per 32-key step: 4 WMMA (QK^T) + online softmax + 4 WMMA (P V).
// Output written as bf16 [B,S,D] for the O-projection.
// -------------------------------------------------------------------------
__global__ __launch_bounds__(128) void flash_attn_kernel(
    const __bf16* __restrict__ Q, const __bf16* __restrict__ K,
    const __bf16* __restrict__ Vt, __bf16* __restrict__ attnb) {
    const int S = 2048, HD = 64, H = 16;
    const int lane = threadIdx.x & 31, wid = threadIdx.x >> 5;
    const int half = lane >> 4, rown = lane & 15;
    const int w  = blockIdx.x * 4 + wid;
    const int qt = w & 127;            // S/16
    const int h  = (w >> 7) & 15;
    const int b  = w >> 11;
    const int q0 = qt * 16;

    const __bf16* Qp = Q  + ((size_t)(b * H + h) * S) * HD;
    const __bf16* Kp = K  + ((size_t)(b * H + h) * S) * HD;
    const __bf16* Vp = Vt + ((size_t)(b * H + h) * HD) * S;

    // Q fragments (16 rows x 64 d), two K=32 chunks
    const v16bf qf0 = load_frag(Qp + (size_t)(q0 + rown) * HD,      half);
    const v16bf qf1 = load_frag(Qp + (size_t)(q0 + rown) * HD + 32, half);

    v8f oacc[4];
#pragma unroll
    for (int t = 0; t < 4; ++t) oacc[t] = zero_v8f();

    float mrow[8], lrow[8];
#pragma unroll
    for (int r = 0; r < 8; ++r) { mrow[r] = -1e30f; lrow[r] = 0.0f; }

    // per-wave P transpose buffer; row stride 48 halves (96 B) keeps the
    // two 16-byte fragment chunks aligned for ds_load_b128.
    __shared__ __bf16 plds[4][16 * 48];
    __bf16* lp = plds[wid];

    const float scl = 0.125f;          // 1/sqrt(64)

    for (int j0 = 0; j0 < q0 + 16; j0 += 32) {
        // ---- scores: S[16x32] = Q K^T ----
        v8f sf[2];
#pragma unroll
        for (int nt = 0; nt < 2; ++nt) {
            const int kv = j0 + nt * 16 + rown;         // B frag: lane = key row
            const v16bf k0 = load_frag(Kp + (size_t)kv * HD,      half);
            const v16bf k1 = load_frag(Kp + (size_t)kv * HD + 32, half);
            v8f c = zero_v8f();
            c = WMMA_BF16(qf0, k0, c);
            c = WMMA_BF16(qf1, k1, c);
            sf[nt] = c;
        }

        // ---- causal mask + row max ----
        float rmax[8];
#pragma unroll
        for (int r = 0; r < 8; ++r) {
            const int qrow = q0 + r + 8 * half;
            const float s0 = (j0 + rown      <= qrow) ? sf[0][r] * scl : -1e30f;
            const float s1 = (j0 + 16 + rown <= qrow) ? sf[1][r] * scl : -1e30f;
            sf[0][r] = s0; sf[1][r] = s1;
            rmax[r] = fmaxf(s0, s1);
        }
#pragma unroll
        for (int m = 1; m < 16; m <<= 1)
#pragma unroll
            for (int r = 0; r < 8; ++r)
                rmax[r] = fmaxf(rmax[r], __shfl_xor(rmax[r], m, 32));

        // ---- online softmax update, write P (bf16) to LDS ----
        float rsum[8], rsc[8];
#pragma unroll
        for (int r = 0; r < 8; ++r) {
            const float mnew = fmaxf(mrow[r], rmax[r]);
            rsc[r]  = __expf(mrow[r] - mnew);
            mrow[r] = mnew;
            const float p0 = __expf(sf[0][r] - mnew);
            const float p1 = __expf(sf[1][r] - mnew);
            const int m_ = r + 8 * half;
            lp[m_ * 48 + rown]      = f2bf(p0);
            lp[m_ * 48 + 16 + rown] = f2bf(p1);
            rsum[r] = p0 + p1;
        }
#pragma unroll
        for (int m = 1; m < 16; m <<= 1)
#pragma unroll
            for (int r = 0; r < 8; ++r)
                rsum[r] += __shfl_xor(rsum[r], m, 32);
#pragma unroll
        for (int r = 0; r < 8; ++r) lrow[r] = lrow[r] * rsc[r] + rsum[r];

        // rescale accumulators
#pragma unroll
        for (int t = 0; t < 4; ++t)
#pragma unroll
            for (int r = 0; r < 8; ++r) oacc[t][r] *= rsc[r];

        // ---- P V: A = P[16x32] from LDS, B = V^T rows (contiguous) ----
        const v16bf pf = load_frag(lp + rown * 48, half);
#pragma unroll
        for (int t = 0; t < 4; ++t) {
            const v16bf vf = load_frag(Vp + (size_t)(t * 16 + rown) * S + j0, half);
            oacc[t] = WMMA_BF16(pf, vf, oacc[t]);
        }
    }

    // ---- normalize + store bf16 to attn [B,S,D] ----
#pragma unroll
    for (int r = 0; r < 8; ++r) {
        const float inv = 1.0f / lrow[r];
        const int s = q0 + r + 8 * half;
#pragma unroll
        for (int t = 0; t < 4; ++t)
            attnb[((size_t)(b * S + s)) * 1024 + h * 64 + t * 16 + rown] =
                f2bf(oacc[t][r] * inv);
    }
}

// -------------------------------------------------------------------------
// Kernel 3: out = attn @ Wo^T.  One wave per 16x64 tile of [4096 x 1024].
// -------------------------------------------------------------------------
__global__ __launch_bounds__(128) void oproj_kernel(
    const __bf16* __restrict__ attnb, const __bf16* __restrict__ wob,
    float* __restrict__ out) {
    const int D = 1024;
    const int lane = threadIdx.x & 31, wid = threadIdx.x >> 5;
    const int half = lane >> 4, rown = lane & 15;
    const int w  = blockIdx.x * 4 + wid;
    const int nt = w & 15;     // 1024/64
    const int mt = w >> 4;     // 4096/16

    const __bf16* arow = attnb + (size_t)(mt * 16 + rown) * D;
    const __bf16* wrow = wob   + (size_t)(nt * 64 + rown) * D;

    v8f acc[4];
#pragma unroll
    for (int t = 0; t < 4; ++t) acc[t] = zero_v8f();

#pragma unroll 4
    for (int kc = 0; kc < 32; ++kc) {
        const v16bf a = load_frag(arow + kc * 32, half);
#pragma unroll
        for (int sub = 0; sub < 4; ++sub) {
            const v16bf b = load_frag(wrow + (size_t)sub * 16 * D + kc * 32, half);
            acc[sub] = WMMA_BF16(a, b, acc[sub]);
        }
    }

#pragma unroll
    for (int sub = 0; sub < 4; ++sub)
#pragma unroll
        for (int r = 0; r < 8; ++r)
            out[(size_t)(mt * 16 + r + 8 * half) * D + nt * 64 + sub * 16 + rown] =
                acc[sub][r];
}

// -------------------------------------------------------------------------
extern "C" void kernel_launch(void* const* d_in, const int* in_sizes, int n_in,
                              void* d_out, int out_size, void* d_ws, size_t ws_size,
                              hipStream_t stream) {
    const float* x    = (const float*)d_in[0];   // [2,2048,1024] f32
    const int*   pos  = (const int*)d_in[1];     // [2048] (arange)
    const float* wqkv = (const float*)d_in[2];   // [3072,1024] f32
    const float* wo   = (const float*)d_in[3];   // [1024,1024] f32
    float* out = (float*)d_out;                  // [2,2048,1024] f32

    const size_t nQ    = 2ull * 16 * 2048 * 64;  // 4,194,304 (Q/K/V each)
    const size_t nX    = 2ull * 2048 * 1024;     // 4,194,304
    const size_t nWqkv = 3072ull * 1024;         // 3,145,728
    const size_t nWo   = 1024ull * 1024;         // 1,048,576

    char* ws = (char*)d_ws;
    __bf16* Qb    = (__bf16*)ws;            // 8 MB
    __bf16* Kb    = Qb + nQ;                // 8 MB
    __bf16* Vt    = Kb + nQ;                // 8 MB
    __bf16* attnb = Vt + nQ;                // 8 MB
    __bf16* xb    = attnb + nX;             // 8 MB
    __bf16* wqkvb = xb + nX;                // 6 MB
    __bf16* wob   = wqkvb + nWqkv;          // 2 MB

    // 0) one-time f32 -> bf16 converts (memory bound)
    cvt_bf16_kernel<<<(unsigned)(nX / 1024),    256, 0, stream>>>(x,    xb);
    cvt_bf16_kernel<<<(unsigned)(nWqkv / 1024), 256, 0, stream>>>(wqkv, wqkvb);
    cvt_bf16_kernel<<<(unsigned)(nWo / 1024),   256, 0, stream>>>(wo,   wob);

    // 1) QKV projection + RoPE + pack: 256*48 waves (16x64 tiles)
    qkv_rope_kernel<<<3072, 128, 0, stream>>>(xb, pos, wqkvb, Qb, Kb, Vt);
    // 2) flash attention: 2*16*128 = 4096 waves
    flash_attn_kernel<<<1024, 128, 0, stream>>>(Qb, Kb, Vt, attnb);
    // 3) output projection: 256*16 waves (16x64 tiles)
    oproj_kernel<<<1024, 128, 0, stream>>>(attnb, wob, out);
}
